// match_lstm_71734543778346
// MI455X (gfx1250) — matrix operand order repaired and checked by reading, not compiled, not run on hardware
//
#include <hip/hip_runtime.h>
#include <hip/hip_bf16.h>

typedef __attribute__((ext_vector_type(16))) __bf16 v16bf;
typedef __attribute__((ext_vector_type(8)))  float  v8f;

#define B_ 128
#define T_ 400
#define Q_ 60
#define H_ 256

union AF { v16bf v; unsigned u[8]; };
union CF { v8f  v; float    f[8]; };

static __device__ __forceinline__ unsigned short f2bf(float x) {
  union { float f; unsigned u; } t; t.f = x;
  unsigned r = t.u + 0x7FFFu + ((t.u >> 16) & 1u);
  return (unsigned short)(r >> 16);
}
static __device__ __forceinline__ float sigf(float x) {
  return 1.0f / (1.0f + __expf(-x));
}
static __device__ __forceinline__ void setAF(AF& a, uint4 x0, uint4 x1) {
  a.u[0] = x0.x; a.u[1] = x0.y; a.u[2] = x0.z; a.u[3] = x0.w;
  a.u[4] = x1.x; a.u[5] = x1.y; a.u[6] = x1.z; a.u[7] = x1.w;
}
// A-fragment: two 16B loads at +0 / +32 bytes (k pairs 0..7 and 16..23).
static __device__ __forceinline__ void loadA64(AF& a, const char* arow, int off) {
  const uint4* ap = (const uint4*)(arow + off);
  setAF(a, ap[0], ap[2]);
}
// B-fragment: one contiguous 32B per lane from the packed buffer.
static __device__ __forceinline__ void loadB(AF& b, const char* bbase, int off) {
  const uint4* bp = (const uint4*)(bbase + off);
  setAF(b, bp[0], bp[1]);
}
// Low 32 bits of a generic LDS pointer are the wave-relative LDS byte address.
static __device__ __forceinline__ unsigned ldsAddr(const void* p) {
  return (unsigned)(size_t)p;
}
// CDNA5 async copy global->LDS, 16B per lane, tracked by ASYNCcnt.
static __device__ __forceinline__ void asyncCopy16(unsigned ldsOff, const void* g) {
  asm volatile("global_load_async_to_lds_b128 %0, %1, off"
               :: "v"(ldsOff), "v"((unsigned long long)(size_t)g)
               : "memory");
}
static __device__ __forceinline__ void asyncWait0() {
  asm volatile("s_wait_asynccnt 0x0" ::: "memory");
}
#define WMMA_BF16(acc, a, b)                                              \
  (acc).v = __builtin_amdgcn_wmma_f32_16x16x32_bf16(                      \
      false, (a).v, false, (b).v, (short)0, (acc).v, false, false)

// ---------------- utility kernels (run once per launch) ----------------

__global__ void k_zero(unsigned* p, int n) {
  int i = blockIdx.x * blockDim.x + threadIdx.x;
  if (i < n) p[i] = 0u;
}

__global__ void k_cvt(const float* s, unsigned short* d, int n) {
  int i = blockIdx.x * blockDim.x + threadIdx.x;
  if (i < n) d[i] = f2bf(s[i]);
}

// Pack row-major fp32 weight W (Ntot x Ktot), computing x @ W^T, into WMMA
// B-fragment layout: fragment (kt,nt) = 32 lanes x 32B; lane holds n = nt*16 +
// (lane&15), k = kt*32 + (lane>>4)*8 + (e&7) + (e>>3)*16, e = 0..15.
__global__ void k_pack(const float* W, unsigned short* dst, int Kt, int Nt, int Ktot) {
  int tid = blockIdx.x * blockDim.x + threadIdx.x;
  int lane = tid & 31;
  int frag = tid >> 5;
  if (frag >= Kt * Nt) return;
  int kt = frag / Nt, nt = frag % Nt;
  int n = nt * 16 + (lane & 15);
  int kb = kt * 32 + ((lane >> 4) << 3);
  unsigned short* o = dst + (size_t)frag * 512 + lane * 16;
#pragma unroll
  for (int e = 0; e < 16; e++)
    o[e] = f2bf(W[(size_t)n * Ktot + kb + (e & 7) + ((e >> 3) << 4)]);
}

// Pack concat(W_ih[:, :2H], W_hh) along K: logical Ktot = 768, Kt=24, Nt=64.
__global__ void k_pack_lstm(const float* Wih, const float* Whh, unsigned short* dst) {
  const int Kt = 24, Nt = 64;
  int tid = blockIdx.x * blockDim.x + threadIdx.x;
  int lane = tid & 31;
  int frag = tid >> 5;
  if (frag >= Kt * Nt) return;
  int kt = frag / Nt, nt = frag % Nt;
  int n = nt * 16 + (lane & 15);
  int kb = kt * 32 + ((lane >> 4) << 3);
  unsigned short* o = dst + (size_t)frag * 512 + lane * 16;
#pragma unroll
  for (int e = 0; e < 16; e++) {
    int k = kb + (e & 7) + ((e >> 3) << 4);
    float w = (k < 512) ? Wih[(size_t)n * 512 + k] : Whh[(size_t)n * 256 + (k - 512)];
    o[e] = f2bf(w);
  }
}

// bf16 WMMA GEMM (N=256, K=256): D[m,n] = sum_k A[m,k]*W[n,k] + bias[n].
// One wave per 16x64 output strip (4 N-tiles, 4 independent accumulators);
// explicit ping-pong pipeline: k-tile kt+1's loads issued before kt's WMMAs.
__global__ void k_gemm(const unsigned short* A, int lda, const unsigned* Bp,
                       const float* bias, float* D, int ldd, int Mt) {
  const int Nt = 16, Kt = 8;
  int wid = (blockIdx.x * blockDim.x + threadIdx.x) >> 5;
  int lane = threadIdx.x & 31;
  if (wid >= Mt * 4) return;
  int mt = wid >> 2, nt0 = (wid & 3) * 4;
  int la = lane & 15, kh = lane >> 4;
  const char* Arow = (const char*)(A + (size_t)(mt * 16 + la) * lda) + kh * 16;
  const char* Bbase = (const char*)Bp + (size_t)nt0 * 1024 + lane * 32;
  CF acc[4];
#pragma unroll
  for (int q = 0; q < 4; q++)
#pragma unroll
    for (int j = 0; j < 8; j++) acc[q].f[j] = 0.0f;

  AF a0, a1, b0[4], b1[4];
  loadA64(a0, Arow, 0);
#pragma unroll
  for (int q = 0; q < 4; q++) loadB(b0[q], Bbase, q * 1024);
#pragma unroll
  for (int kt = 0; kt < Kt; kt += 2) {
    loadA64(a1, Arow, (kt + 1) * 64);
#pragma unroll
    for (int q = 0; q < 4; q++)
      loadB(b1[q], Bbase, (kt + 1) * Nt * 1024 + q * 1024);
#pragma unroll
    for (int q = 0; q < 4; q++) WMMA_BF16(acc[q], a0, b0[q]);
    if (kt + 2 < Kt) {
      loadA64(a0, Arow, (kt + 2) * 64);
#pragma unroll
      for (int q = 0; q < 4; q++)
        loadB(b0[q], Bbase, (kt + 2) * Nt * 1024 + q * 1024);
    }
#pragma unroll
    for (int q = 0; q < 4; q++) WMMA_BF16(acc[q], a1, b1[q]);
  }
#pragma unroll
  for (int q = 0; q < 4; q++) {
    int n = (nt0 + q) * 16 + la;
    float bv = bias ? bias[n] : 0.0f;
#pragma unroll
    for (int j = 0; j < 8; j++) {
      int m = mt * 16 + j + kh * 8;
      D[(size_t)m * ldd + n] = acc[q].f[j] + bv;
    }
  }
}

// ---------------- per-timestep kernels ----------------
// X layout per (dir,b): [ctx(0:256) | af(256:512) | h(512:768)] bf16.

// K1a: s[dir][b][:] = Pre2[b, t_eff] + h_bf16 @ W_g3^T.
// 8 waves/block share one 16-row h tile, async-staged into LDS.
__global__ void k1a_sgemm(const unsigned short* X, const unsigned* Bp_g3,
                          const float* Pre2, float* sbuf, int t) {
  __shared__ __align__(16) unsigned short tileA[16 * 256];  // 8 KB
  int tid = threadIdx.x;
  int wid = blockIdx.x * 8 + (tid >> 5);
  int lane = tid & 31;
  int dir = wid >> 7, r = wid & 127;
  int mt = r >> 4, nt = r & 15;  // mt, dir uniform per block
  int la = lane & 15, kh = lane >> 4;
  int t_eff = dir ? (T_ - 1 - t) : t;

  // async-stage h-part of 16 rows (16 x 512B) into LDS
  const char* src = (const char*)(X + (size_t)(dir * B_ + mt * 16) * 768 + 512);
  unsigned lbase = ldsAddr(tileA);
#pragma unroll
  for (int i = 0; i < 2; i++) {
    int idx = i * 256 + tid;
    int row = idx >> 5, c = (idx & 31) * 16;
    asyncCopy16(lbase + row * 512 + c, src + row * 1536 + c);
  }
  asyncWait0();
  __syncthreads();

  int n = nt * 16 + la;
  CF acc;
#pragma unroll
  for (int j = 0; j < 8; j++) {
    int b = mt * 16 + j + kh * 8;
    acc.f[j] = Pre2[((size_t)b * T_ + t_eff) * H_ + n];
  }
  const char* arow = (const char*)tileA + la * 512 + kh * 16;
  const char* bbase = (const char*)Bp_g3 + (size_t)nt * 1024 + lane * 32;
  AF a0, a1, b0, b1;
  loadA64(a0, arow, 0);
  loadB(b0, bbase, 0);
#pragma unroll
  for (int kt = 0; kt < 8; kt += 2) {
    loadA64(a1, arow, (kt + 1) * 64);
    loadB(b1, bbase, (size_t)(kt + 1) * 16 * 1024);
    WMMA_BF16(acc, a0, b0);
    if (kt + 2 < 8) {
      loadA64(a0, arow, (kt + 2) * 64);
      loadB(b0, bbase, (size_t)(kt + 2) * 16 * 1024);
    }
    WMMA_BF16(acc, a1, b1);
  }
#pragma unroll
  for (int j = 0; j < 8; j++) {
    int b = mt * 16 + j + kh * 8;
    sbuf[((size_t)dir * B_ + b) * H_ + n] = acc.f[j];
  }
}

// K1b: attention (tanh + softmax + weighted sum); writes af (bf16) + ctx row into X.
__global__ void k1b_attend(unsigned short* X, const unsigned short* ctxb,
                           const float* sbuf, const float* Pg1, const float* qh,
                           const float* Wg4, const float* bg4, int t) {
  __shared__ float s_sh[H_];
  __shared__ float e_sh[64];
  int dir = blockIdx.x >> 7;
  int b = blockIdx.x & 127;
  int tid = threadIdx.x;
  int t_eff = dir ? (T_ - 1 - t) : t;
  X[(size_t)(dir * B_ + b) * 768 + tid] = ctxb[((size_t)b * T_ + t_eff) * H_ + tid];
  s_sh[tid] = sbuf[((size_t)dir * B_ + b) * H_ + tid];
  __syncthreads();
  int w = tid >> 5, lane = tid & 31;
  for (int q = w; q < Q_; q += 8) {
    const float* pg = Pg1 + ((size_t)b * Q_ + q) * H_;
    float sum = 0.0f;
    for (int k = lane; k < H_; k += 32)
      sum += Wg4[k] * tanhf(s_sh[k] + pg[k]);
#pragma unroll
    for (int off = 16; off; off >>= 1) sum += __shfl_xor(sum, off, 32);
    if (lane == 0) e_sh[q] = sum + bg4[0];
  }
  __syncthreads();
  if (tid < 32) {
    float e1 = (tid < Q_) ? e_sh[tid] : -3.4e38f;
    float e2 = (tid + 32 < Q_) ? e_sh[tid + 32] : -3.4e38f;
    float m = fmaxf(e1, e2);
#pragma unroll
    for (int off = 16; off; off >>= 1) m = fmaxf(m, __shfl_xor(m, off, 32));
    float x1 = (tid < Q_) ? __expf(e1 - m) : 0.0f;
    float x2 = (tid + 32 < Q_) ? __expf(e2 - m) : 0.0f;
    float s = x1 + x2;
#pragma unroll
    for (int off = 16; off; off >>= 1) s += __shfl_xor(s, off, 32);
    float inv = 1.0f / s;
    if (tid < Q_) e_sh[tid] = x1 * inv;
    if (tid + 32 < Q_) e_sh[tid + 32] = x2 * inv;
  }
  __syncthreads();
  float af = 0.0f;
  const float* qrow = qh + (size_t)b * Q_ * H_ + tid;
  for (int q = 0; q < Q_; q++) af += e_sh[q] * qrow[(size_t)q * H_];
  X[(size_t)(dir * B_ + b) * 768 + 256 + tid] = f2bf(af);
}

// K2: gates = [ctx|af|h] @ Wcat^T (+bias), fused LSTM pointwise update.
// 8 waves/block share one 16-row X tile (24 KB) async-staged into LDS;
// each wave owns one (batch-tile, hidden-tile) and all 4 gates (4 accums).
// Ping-pong pipeline on the packed-weight stream (96 KB/wave from L2).
__global__ void k2_lstm(unsigned short* X, const unsigned* Bp_f, const unsigned* Bp_r,
                        const float* bih_f, const float* bhh_f,
                        const float* bih_r, const float* bhh_r,
                        float* cst, float* out, int t) {
  __shared__ __align__(16) unsigned short tileA[16 * 768];  // 24 KB
  int tid = threadIdx.x;
  int wid = blockIdx.x * 8 + (tid >> 5);
  int lane = tid & 31;
  int dir = wid >> 7, r = wid & 127;
  int mt = r >> 4, hnt = r & 15;  // mt, dir uniform per block
  int la = lane & 15, kh = lane >> 4;

  // async-stage the full 16-row X tile (contiguous 24576 B) into LDS
  const char* src = (const char*)(X + (size_t)(dir * B_ + mt * 16) * 768);
  unsigned lbase = ldsAddr(tileA);
#pragma unroll
  for (int i = 0; i < 6; i++) {
    int o = (i * 256 + tid) * 16;
    asyncCopy16(lbase + o, src + o);
  }
  asyncWait0();
  __syncthreads();

  const unsigned* Bp = dir ? Bp_r : Bp_f;
  const char* arow = (const char*)tileA + la * 1536 + kh * 16;
  const char* bbase = (const char*)Bp + (size_t)hnt * 1024 + lane * 32;
  CF acc[4];
#pragma unroll
  for (int g = 0; g < 4; g++)
#pragma unroll
    for (int j = 0; j < 8; j++) acc[g].f[j] = 0.0f;

  AF a0, a1, b0[4], b1[4];
  loadA64(a0, arow, 0);
#pragma unroll
  for (int g = 0; g < 4; g++) loadB(b0[g], bbase, g * 16 * 1024);
#pragma unroll
  for (int kt = 0; kt < 24; kt += 2) {
    loadA64(a1, arow, (kt + 1) * 64);
#pragma unroll
    for (int g = 0; g < 4; g++)
      loadB(b1[g], bbase, (size_t)(kt + 1) * 64 * 1024 + (size_t)g * 16 * 1024);
#pragma unroll
    for (int g = 0; g < 4; g++) WMMA_BF16(acc[g], a0, b0[g]);
    if (kt + 2 < 24) {
      loadA64(a0, arow, (kt + 2) * 64);
#pragma unroll
      for (int g = 0; g < 4; g++)
        loadB(b0[g], bbase, (size_t)(kt + 2) * 64 * 1024 + (size_t)g * 16 * 1024);
    }
#pragma unroll
    for (int g = 0; g < 4; g++) WMMA_BF16(acc[g], a1, b1[g]);
  }

  int hidx = hnt * 16 + la;
  const float* bih = dir ? bih_r : bih_f;
  const float* bhh = dir ? bhh_r : bhh_f;
  float bs0 = bih[hidx]       + bhh[hidx];
  float bs1 = bih[256 + hidx] + bhh[256 + hidx];
  float bs2 = bih[512 + hidx] + bhh[512 + hidx];
  float bs3 = bih[768 + hidx] + bhh[768 + hidx];
  int t_out = dir ? (T_ - 1 - t) : t;
#pragma unroll
  for (int j = 0; j < 8; j++) {
    int b = mt * 16 + j + kh * 8;
    float iv = sigf(acc[0].f[j] + bs0);
    float fv = sigf(acc[1].f[j] + bs1);
    float gv = tanhf(acc[2].f[j] + bs2);
    float ov = sigf(acc[3].f[j] + bs3);
    float* cp = cst + ((size_t)dir * B_ + b) * H_ + hidx;
    float cn = fv * cp[0] + iv * gv;
    cp[0] = cn;
    float hn = ov * tanhf(cn);
    X[(size_t)(dir * B_ + b) * 768 + 512 + hidx] = f2bf(hn);
    out[((size_t)b * T_ + t_out) * (2 * H_) + dir * H_ + hidx] = hn;
  }
}

// ---------------- host ----------------

extern "C" void kernel_launch(void* const* d_in, const int* in_sizes, int n_in,
                              void* d_out, int out_size, void* d_ws, size_t ws_size,
                              hipStream_t stream) {
  const float* ctx   = (const float*)d_in[0];
  const float* qh    = (const float*)d_in[1];
  const float* Wg1   = (const float*)d_in[2];
  const float* Wg2   = (const float*)d_in[3];
  const float* bg2   = (const float*)d_in[4];
  const float* Wg3   = (const float*)d_in[5];
  const float* Wg4   = (const float*)d_in[6];
  const float* bg4   = (const float*)d_in[7];
  const float* Wih_f = (const float*)d_in[8];
  const float* Whh_f = (const float*)d_in[9];
  const float* bih_f = (const float*)d_in[10];
  const float* bhh_f = (const float*)d_in[11];
  const float* Wih_r = (const float*)d_in[12];
  const float* Whh_r = (const float*)d_in[13];
  const float* bih_r = (const float*)d_in[14];
  const float* bhh_r = (const float*)d_in[15];
  float* out = (float*)d_out;

  char* ws = (char*)d_ws;
  size_t off = 0;
  auto alloc = [&](size_t bytes) -> void* {
    void* p = ws + off;
    off = (off + bytes + 255) & ~(size_t)255;
    return p;
  };
  unsigned short* ctxb = (unsigned short*)alloc((size_t)B_ * T_ * H_ * 2);  // 26.2 MB
  unsigned short* qhb  = (unsigned short*)alloc((size_t)B_ * Q_ * H_ * 2);  //  3.9 MB
  float* Pre2 = (float*)alloc((size_t)B_ * T_ * H_ * 4);                    // 52.4 MB
  float* Pg1  = (float*)alloc((size_t)B_ * Q_ * H_ * 4);                    //  7.9 MB
  unsigned* Bp_g1 = (unsigned*)alloc(8 * 16 * 1024);
  unsigned* Bp_g2 = (unsigned*)alloc(8 * 16 * 1024);
  unsigned* Bp_g3 = (unsigned*)alloc(8 * 16 * 1024);
  unsigned* Bp_f  = (unsigned*)alloc(24 * 64 * 1024);
  unsigned* Bp_r  = (unsigned*)alloc(24 * 64 * 1024);
  unsigned short* X = (unsigned short*)alloc((size_t)2 * B_ * 768 * 2);     // 393 KB
  float* sbuf = (float*)alloc((size_t)2 * B_ * H_ * 4);
  float* cst  = (float*)alloc((size_t)2 * B_ * H_ * 4);

  // zero X + sbuf + cst (contiguous, sizes all multiples of 256B)
  int zero_words = (393216 + 262144 + 262144) / 4;
  k_zero<<<(zero_words + 255) / 256, 256, 0, stream>>>((unsigned*)X, zero_words);

  // fp32 -> bf16 conversions
  k_cvt<<<(B_ * T_ * H_ + 255) / 256, 256, 0, stream>>>(ctx, ctxb, B_ * T_ * H_);
  k_cvt<<<(B_ * Q_ * H_ + 255) / 256, 256, 0, stream>>>(qh, qhb, B_ * Q_ * H_);

  // pack weights into WMMA B-fragment layout (stays hot in 192MB L2)
  k_pack<<<16, 256, 0, stream>>>(Wg1, (unsigned short*)Bp_g1, 8, 16, 256);
  k_pack<<<16, 256, 0, stream>>>(Wg2, (unsigned short*)Bp_g2, 8, 16, 256);
  k_pack<<<16, 256, 0, stream>>>(Wg3, (unsigned short*)Bp_g3, 8, 16, 256);
  k_pack_lstm<<<192, 256, 0, stream>>>(Wih_f, Whh_f, (unsigned short*)Bp_f);
  k_pack_lstm<<<192, 256, 0, stream>>>(Wih_r, Whh_r, (unsigned short*)Bp_r);

  // big parallel GEMMs: Pg1 = qh @ Wg1^T ; Pre2 = ctx @ Wg2^T + b_g2
  // waves = Mt * 4 (one wave per 16x64 strip)
  k_gemm<<<240, 256, 0, stream>>>(qhb, 256, Bp_g1, nullptr, Pg1, 256, 480);
  k_gemm<<<1600, 256, 0, stream>>>(ctxb, 256, Bp_g2, bg2, Pre2, 256, 3200);

  // sequential scan: 400 steps, both directions per launch
  for (int t = 0; t < T_; t++) {
    k1a_sgemm<<<32, 256, 0, stream>>>(X, Bp_g3, Pre2, sbuf, t);
    k1b_attend<<<256, 256, 0, stream>>>(X, ctxb, sbuf, Pg1, qh, Wg4, bg4, t);
    k2_lstm<<<32, 256, 0, stream>>>(X, Bp_f, Bp_r, bih_f, bhh_f, bih_r, bhh_r,
                                    cst, out, t);
  }
}
